// DifferentiableQuantumEmbedding_60902636257658
// MI455X (gfx1250) — matrix operand order repaired and checked by reading, not compiled, not run on hardware
//
#include <hip/hip_runtime.h>
#include <hip/hip_bf16.h>
#include <cstddef>

// ---------------------------------------------------------------------------
// DifferentiableQuantumEmbedding for MI455X (gfx1250, wave32, WMMA)
//
//   out = LN(x) |> 0.5*sin(2h) |> Re(FFT16 along qubit axis) |> Dense(W,b)
//
// Re(FFT16) is linear -> folded into the dense weight once per launch:
//   W''[i][n] = sum_q cos(pi/8 * q*qi) * W[q*64 + d][n],  i = qi*64 + d
// leaving a single memory-bound 16384 x 1024 x 1024 GEMM that we run with
// v_wmma_f32_16x16x32_bf16 (fp32 accumulation, bf16 staged operands).
// ---------------------------------------------------------------------------

#define HIDDEN   1024
#define TOKENS   16384          // B*S = 4*4096
#define NQ       16
#define QD       64

typedef __attribute__((ext_vector_type(16))) __bf16 v16bf;
typedef __attribute__((ext_vector_type(8)))  __bf16 v8bf;
typedef __attribute__((ext_vector_type(4)))  __bf16 v4bf;
typedef __attribute__((ext_vector_type(8)))  float  v8f;

// ---------------------------------------------------------------------------
// K1: fold Re(DFT16) into the dense weight; store W''^T (row n = out column,
//     contiguous in K) as bf16. 1M elems x 16 MACs -> negligible.
// ---------------------------------------------------------------------------
__global__ __launch_bounds__(256)
void qe_fold_weights(const float* __restrict__ W, unsigned short* __restrict__ wT_raw) {
    __bf16* wT = reinterpret_cast<__bf16*>(wT_raw);
    int idx = blockIdx.x * 256 + threadIdx.x;        // 0 .. 1024*1024-1
    int n = idx >> 10;                               // output column
    int i = idx & (HIDDEN - 1);                      // input (K) index
    int qi = i >> 6;
    int d  = i & (QD - 1);
    float s = 0.0f;
#pragma unroll
    for (int q = 0; q < NQ; ++q) {
        // cos(2*pi*q*qi/16), phase reduced mod 16 before the fast cos
        float c = __cosf(0.39269908169872414f * (float)((q * qi) & 15));
        s += c * W[(size_t)(q * QD + d) * HIDDEN + n];
    }
    wT[(size_t)n * HIDDEN + i] = (__bf16)s;
}

// ---------------------------------------------------------------------------
// K2: one wave32 per token: LayerNorm (shuffle reduction) + 0.5*sin(2h),
//     emit bf16 activation row. 8 waves per 256-thread block.
// ---------------------------------------------------------------------------
__global__ __launch_bounds__(256)
void qe_act_prep(const float* __restrict__ x,
                 const float* __restrict__ ln_scale,
                 const float* __restrict__ ln_bias,
                 unsigned short* __restrict__ A_raw) {
    __bf16* A = reinterpret_cast<__bf16*>(A_raw);
    const int lane = threadIdx.x & 31;
    const int t    = blockIdx.x * 8 + (threadIdx.x >> 5);

    const float* xr = x + (size_t)t * HIDDEN;
    float4 v[8];
    float s = 0.0f, sq = 0.0f;
#pragma unroll
    for (int j = 0; j < 8; ++j) {
        v[j] = *(const float4*)(xr + j * 128 + lane * 4);
        s  += v[j].x + v[j].y + v[j].z + v[j].w;
        sq += v[j].x * v[j].x + v[j].y * v[j].y + v[j].z * v[j].z + v[j].w * v[j].w;
    }
#pragma unroll
    for (int off = 16; off >= 1; off >>= 1) {
        s  += __shfl_xor(s,  off, 32);
        sq += __shfl_xor(sq, off, 32);
    }
    const float mean = s * (1.0f / HIDDEN);
    const float var  = sq * (1.0f / HIDDEN) - mean * mean;
    const float inv  = rsqrtf(var + 1e-6f);

    __bf16* ar = A + (size_t)t * HIDDEN;
#pragma unroll
    for (int j = 0; j < 8; ++j) {
        const int e0 = j * 128 + lane * 4;
        float4 g = *(const float4*)(ln_scale + e0);
        float4 b = *(const float4*)(ln_bias  + e0);
        float h0 = (v[j].x - mean) * inv * g.x + b.x;
        float h1 = (v[j].y - mean) * inv * g.y + b.y;
        float h2 = (v[j].z - mean) * inv * g.z + b.z;
        float h3 = (v[j].w - mean) * inv * g.w + b.w;
        v4bf o;
        o[0] = (__bf16)(0.5f * __sinf(2.0f * h0));
        o[1] = (__bf16)(0.5f * __sinf(2.0f * h1));
        o[2] = (__bf16)(0.5f * __sinf(2.0f * h2));
        o[3] = (__bf16)(0.5f * __sinf(2.0f * h3));
        *(v4bf*)(ar + e0) = o;
    }
}

// ---------------------------------------------------------------------------
// K3: GEMM  out[16384,1024] = A[16384,1024](bf16) x W''[1024,1024] + bias
//     Block: 256 threads = 8 waves (2 M x 4 N), block tile 128x128,
//     wave tile 64x32 = 4x2 v_wmma_f32_16x16x32_bf16 accumulators,
//     K staged in 32-wide LDS tiles.
// ---------------------------------------------------------------------------
#define LDS_PITCH 40   // 32 + 8 half pad (80B rows: keeps 16B alignment)

__global__ __launch_bounds__(256)
void qe_gemm(const unsigned short* __restrict__ A_raw,
             const unsigned short* __restrict__ wT_raw,
             const float* __restrict__ bias,
             float* __restrict__ out) {
    const __bf16* A  = reinterpret_cast<const __bf16*>(A_raw);
    const __bf16* wT = reinterpret_cast<const __bf16*>(wT_raw);

    __shared__ __bf16 Atile[128][LDS_PITCH];
    __shared__ __bf16 Btile[128][LDS_PITCH];

    const int tid    = threadIdx.x;
    const int lane   = tid & 31;
    const int wave   = tid >> 5;
    const int wave_m = wave & 1;        // 2 waves along M (64 rows each)
    const int wave_n = wave >> 1;       // 4 waves along N (32 cols each)
    const int row0   = blockIdx.x * 128;
    const int col0   = blockIdx.y * 128;

    const int lhalf = lane >> 4;        // 0: lanes 0-15, 1: lanes 16-31
    const int lmod  = lane & 15;

    v8f acc[4][2];
#pragma unroll
    for (int mt = 0; mt < 4; ++mt)
#pragma unroll
        for (int nt = 0; nt < 2; ++nt)
#pragma unroll
            for (int r = 0; r < 8; ++r) acc[mt][nt][r] = 0.0f;

    for (int k0 = 0; k0 < HIDDEN; k0 += 32) {
        // ---- stage 128x32 A and 128x32 B^T tiles (16B chunks, 2 per thread)
#pragma unroll
        for (int j = 0; j < 2; ++j) {
            const int c  = tid + j * 256;       // 0..511 chunks of 8 halves
            const int r  = c >> 2;
            const int cc = (c & 3) * 8;
            *(uint4*)&Atile[r][cc] =
                *(const uint4*)(A  + (size_t)(row0 + r) * HIDDEN + k0 + cc);
            *(uint4*)&Btile[r][cc] =
                *(const uint4*)(wT + (size_t)(col0 + r) * HIDDEN + k0 + cc);
        }
        __syncthreads();

        // ---- A fragments: lane m = lmod; lanes 0-15 K {0..7,16..23},
        //      lanes 16-31 K {8..15,24..31}   (CDNA5 16-bit A layout)
        v16bf afrag[4];
#pragma unroll
        for (int mt = 0; mt < 4; ++mt) {
            const __bf16* ap = &Atile[wave_m * 64 + mt * 16 + lmod][lhalf * 8];
            v8bf lo = *(const v8bf*)(ap);
            v8bf hi = *(const v8bf*)(ap + 16);
#pragma unroll
            for (int e = 0; e < 8; ++e) { afrag[mt][e] = lo[e]; afrag[mt][e + 8] = hi[e]; }
        }
        // ---- B fragments: lane n = lmod; lanes 0-15 K {0..7,16..23}, 16-31 K {8..15,24..31}
        v16bf bfrag[2];
#pragma unroll
        for (int nt = 0; nt < 2; ++nt) {
            const __bf16* bp = &Btile[wave_n * 32 + nt * 16 + lmod][lhalf * 8];
            v8bf lo = *(const v8bf*)(bp);
            v8bf hi = *(const v8bf*)(bp + 16);
#pragma unroll
            for (int e = 0; e < 8; ++e) { bfrag[nt][e] = lo[e]; bfrag[nt][e + 8] = hi[e]; }
        }

#pragma unroll
        for (int mt = 0; mt < 4; ++mt)
#pragma unroll
            for (int nt = 0; nt < 2; ++nt)
                acc[mt][nt] = __builtin_amdgcn_wmma_f32_16x16x32_bf16(
                    false, afrag[mt], false, bfrag[nt],
                    (short)0, acc[mt][nt], false, false);

        __syncthreads();
    }

    // ---- epilogue: C layout is n = lmod, m = r + 8*lhalf per VGPR r
#pragma unroll
    for (int nt = 0; nt < 2; ++nt) {
        const int n = col0 + wave_n * 32 + nt * 16 + lmod;
        const float bv = bias[n];
#pragma unroll
        for (int mt = 0; mt < 4; ++mt) {
            const int mbase = row0 + wave_m * 64 + mt * 16 + 8 * lhalf;
#pragma unroll
            for (int r = 0; r < 8; ++r)
                out[(size_t)(mbase + r) * HIDDEN + n] = acc[mt][nt][r] + bv;
        }
    }
}

// ---------------------------------------------------------------------------
extern "C" void kernel_launch(void* const* d_in, const int* in_sizes, int n_in,
                              void* d_out, int out_size, void* d_ws, size_t ws_size,
                              hipStream_t stream) {
    (void)in_sizes; (void)n_in; (void)out_size; (void)ws_size;
    const float* x        = (const float*)d_in[0];
    const float* ln_scale = (const float*)d_in[1];
    const float* ln_bias  = (const float*)d_in[2];
    const float* dense_w  = (const float*)d_in[3];
    const float* dense_b  = (const float*)d_in[4];
    float* out = (float*)d_out;

    // workspace: [0,2MB) folded W''^T (bf16), [2MB,34MB) bf16 activations
    unsigned short* wT = (unsigned short*)d_ws;
    unsigned short* Ab = wT + (size_t)HIDDEN * HIDDEN;

    qe_fold_weights<<<(HIDDEN * HIDDEN) / 256, 256, 0, stream>>>(dense_w, wT);
    qe_act_prep<<<TOKENS / 8, 256, 0, stream>>>(x, ln_scale, ln_bias, Ab);

    dim3 grid(TOKENS / 128, HIDDEN / 128);
    qe_gemm<<<grid, 256, 0, stream>>>(Ab, wT, dense_b, out);
}
